// MultiHeadAttentionBlock_8589935279
// MI455X (gfx1250) — matrix operand order repaired and checked by reading, not compiled, not run on hardware
//
#include <hip/hip_runtime.h>
#include <hip/hip_bf16.h>
#include <cstdint>

// ---------------------------------------------------------------------------
// MultiHeadAttentionBlock for MI455X (gfx1250): bf16 WMMA everywhere.
// B=2, S=2048, D=1024, H=16, DK=64. All matmuls use v_wmma_f32_16x16x32_bf16.
// GEMMs: 32x64 register tile / wave, software-pipelined (double-buffered
// fragments) so loads overlap WMMA instead of s_wait_loadcnt 0 per op.
// ---------------------------------------------------------------------------

namespace {
constexpr int kB = 2, kS = 2048, kD = 1024, kH = 16, kDK = 64;
constexpr int kM = kB * kS;  // 4096 rows for the [M,K]x[K,N] projection GEMMs

constexpr size_t SZ_ACT = (size_t)kM * kD * sizeof(unsigned short);  // 8 MiB
constexpr size_t SZ_W   = (size_t)kD * kD * sizeof(unsigned short);  // 2 MiB
constexpr size_t PAD    = 4096;  // slack between regions (allows benign tail overread)

constexpr size_t OFF_QB  = 0;
constexpr size_t OFF_KB  = OFF_QB  + SZ_ACT + PAD;
constexpr size_t OFF_VB  = OFF_KB  + SZ_ACT + PAD;
constexpr size_t OFF_WTQ = OFF_VB  + SZ_ACT + PAD;
constexpr size_t OFF_WTK = OFF_WTQ + SZ_W   + PAD;
constexpr size_t OFF_WTV = OFF_WTK + SZ_W   + PAD;
constexpr size_t OFF_WTO = OFF_WTV + SZ_W   + PAD;
constexpr size_t OFF_QH  = OFF_WTO + SZ_W   + PAD;   // [B,H,S,DK] bf16
constexpr size_t OFF_KH  = OFF_QH  + SZ_ACT + PAD;   // [B,H,S,DK] bf16
constexpr size_t OFF_VT  = OFF_KH  + SZ_ACT + PAD;   // [B,H,DK,S] bf16 (transposed)
constexpr size_t OFF_AO  = OFF_VT  + SZ_ACT + PAD;   // [B,S,D]    bf16 (merged heads)
}  // namespace

typedef __attribute__((ext_vector_type(16))) __bf16 v16bf;
typedef __attribute__((ext_vector_type(8)))  float  v8f;

union FragAB {
  v16bf v;
  uint4 q[2];
};

__device__ __forceinline__ unsigned short f32_to_bf16(float f) {
  unsigned int u = __float_as_uint(f);
  u += 0x7FFFu + ((u >> 16) & 1u);  // round-to-nearest-even
  return (unsigned short)(u >> 16);
}

__device__ __forceinline__ v8f v8f_zero() {
  v8f z;
#pragma unroll
  for (int i = 0; i < 8; ++i) z[i] = 0.0f;
  return z;
}

__device__ __forceinline__ v8f wmma_bf16(const FragAB& a, const FragAB& b, v8f c) {
  // D = A(16x32 bf16) * B(32x16 bf16) + C(16x16 f32)
  return __builtin_amdgcn_wmma_f32_16x16x32_bf16(
      /*neg_a=*/false, a.v, /*neg_b=*/false, b.v,
      /*c_mod=*/(short)0, c, /*reuse_a=*/false, /*reuse_b=*/false);
}

// ---------------------------------------------------------------------------
// Kernel 1: fp32 -> bf16 streaming cast (activations), 4 elements / thread.
// ---------------------------------------------------------------------------
__global__ void mha_cast_bf16(const float* __restrict__ src,
                              unsigned short* __restrict__ dst, int n) {
  int i = (blockIdx.x * blockDim.x + threadIdx.x) * 4;
  if (i + 3 < n) {
    float4 f = *(const float4*)(src + i);
    unsigned int lo = (unsigned)f32_to_bf16(f.x) | ((unsigned)f32_to_bf16(f.y) << 16);
    unsigned int hi = (unsigned)f32_to_bf16(f.z) | ((unsigned)f32_to_bf16(f.w) << 16);
    uint2 o; o.x = lo; o.y = hi;
    *(uint2*)(dst + i) = o;
  }
}

// ---------------------------------------------------------------------------
// Kernel 2: weight transpose + cast:  wt[n][k] = bf16(w[k][n]).
// LDS-tiled 32x32 so both the read and the write are coalesced.
// ---------------------------------------------------------------------------
__global__ void mha_transpose_cast(const float* __restrict__ w,
                                   unsigned short* __restrict__ wt) {
  __shared__ unsigned short tile[32][33];
  const int bx = blockIdx.x & 31;   // column tile
  const int by = blockIdx.x >> 5;   // row tile
  const int tx = threadIdx.x & 31;
  const int ty = threadIdx.x >> 5;  // 0..7
#pragma unroll
  for (int i = 0; i < 4; ++i) {
    int r = by * 32 + ty + i * 8;
    int c = bx * 32 + tx;
    tile[ty + i * 8][tx] = f32_to_bf16(w[(size_t)r * kD + c]);
  }
  __syncthreads();
#pragma unroll
  for (int i = 0; i < 4; ++i) {
    int n = bx * 32 + ty + i * 8;
    int k = by * 32 + tx;
    wt[(size_t)n * kD + k] = tile[tx][ty + i * 8];
  }
}

// ---------------------------------------------------------------------------
// Shared GEMM core: one wave computes a 32x64 tile of A[4096,1024] @ Wt^T,
// software-pipelined over K with double-buffered fragments.
// ---------------------------------------------------------------------------
__device__ __forceinline__ void gemm_load_a(FragAB a[2],
                                            const unsigned short* __restrict__ A,
                                            int m0, int row, int koffA, int k0) {
#pragma unroll
  for (int mi = 0; mi < 2; ++mi) {
    const unsigned short* ap = A + (size_t)(m0 + mi * 16 + row) * kD + k0 + koffA;
    a[mi].q[0] = *(const uint4*)(ap);
    a[mi].q[1] = *(const uint4*)(ap + 8);  // contiguous 32B of K
  }
}

__device__ __forceinline__ void gemm_load_b(FragAB b[4],
                                            const unsigned short* __restrict__ Wt,
                                            int n0, int row, int koffB, int k0) {
#pragma unroll
  for (int j = 0; j < 4; ++j) {
    const unsigned short* bp = Wt + (size_t)(n0 + j * 16 + row) * kD + k0 + koffB;
    b[j].q[0] = *(const uint4*)(bp);
    b[j].q[1] = *(const uint4*)(bp + 8);  // contiguous 32B of K
  }
}

__device__ __forceinline__ void gemm_core_32x64(const unsigned short* __restrict__ A,
                                                const unsigned short* __restrict__ Wt,
                                                int m0, int n0, int lane,
                                                v8f acc[2][4]) {
  const int row   = lane & 15;
  const int koffA = (lane & 16) ? 16 : 0;
  const int koffB = (lane & 16) ? 16 : 0;

#pragma unroll
  for (int mi = 0; mi < 2; ++mi)
#pragma unroll
    for (int j = 0; j < 4; ++j) acc[mi][j] = v8f_zero();

  FragAB a0[2], b0[4], a1[2], b1[4];
  gemm_load_a(a0, A, m0, row, koffA, 0);
  gemm_load_b(b0, Wt, n0, row, koffB, 0);

  for (int k0 = 0; k0 < kD; k0 += 64) {
    // prefetch k0+32 while the k0 WMMAs run
    gemm_load_a(a1, A, m0, row, koffA, k0 + 32);
    gemm_load_b(b1, Wt, n0, row, koffB, k0 + 32);
#pragma unroll
    for (int mi = 0; mi < 2; ++mi)
#pragma unroll
      for (int j = 0; j < 4; ++j)
        acc[mi][j] = wmma_bf16(a0[mi], b0[j], acc[mi][j]);

    if (k0 + 64 < kD) {  // prefetch k0+64 while the k0+32 WMMAs run
      gemm_load_a(a0, A, m0, row, koffA, k0 + 64);
      gemm_load_b(b0, Wt, n0, row, koffB, k0 + 64);
    }
#pragma unroll
    for (int mi = 0; mi < 2; ++mi)
#pragma unroll
      for (int j = 0; j < 4; ++j)
        acc[mi][j] = wmma_bf16(a1[mi], b1[j], acc[mi][j]);
  }
}

// ---------------------------------------------------------------------------
// Kernel 3: projection GEMM  out = A[4096,1024] @ W + b  (bf16 in, bf16 out).
// mode 0/1: store head-split [B,H,S,DK]; mode 2: store transposed [B,H,DK,S].
// ---------------------------------------------------------------------------
__global__ void mha_proj_gemm(const unsigned short* __restrict__ A,
                              const unsigned short* __restrict__ Wt,
                              const float* __restrict__ bias,
                              unsigned short* __restrict__ dst, int mode) {
  const int wave  = (int)((blockIdx.x * blockDim.x + threadIdx.x) >> 5);
  const int lane  = threadIdx.x & 31;
  const int mtile = wave >> 4;       // 0..127  (32-row tiles)
  const int ngrp  = wave & 15;       // 0..15   (64-col groups)
  const int m0 = mtile * 32;
  const int n0 = ngrp * 64;
  const int row   = lane & 15;
  const int rhalf = (lane & 16) ? 8 : 0;

  v8f acc[2][4];
  gemm_core_32x64(A, Wt, m0, n0, lane, acc);

#pragma unroll
  for (int j = 0; j < 4; ++j) {
    const int n  = n0 + j * 16 + row;
    const float bv = bias[n];
    const int h  = n / kDK;
    const int dk = n % kDK;
#pragma unroll
    for (int mi = 0; mi < 2; ++mi)
#pragma unroll
      for (int g = 0; g < 8; ++g) {
        const int m = m0 + mi * 16 + g + rhalf;
        const int bidx = m / kS;
        const int s    = m % kS;
        const unsigned short o = f32_to_bf16(acc[mi][j][g] + bv);
        if (mode == 2)
          dst[(((size_t)bidx * kH + h) * kDK + dk) * kS + s] = o;  // V transposed
        else
          dst[(((size_t)bidx * kH + h) * kS + s) * kDK + dk] = o;  // Q/K head-split
      }
  }
}

// ---------------------------------------------------------------------------
// Kernel 4: causal flash attention. One wave owns a 16-row query tile of one
// (batch, head); iterates kv in 32-wide blocks up to the diagonal.
// All K/V fragment loads for a block are issued up front so their latency
// hides under score WMMAs + softmax VALU work.
// ---------------------------------------------------------------------------
__global__ void mha_attention(const unsigned short* __restrict__ Qh,
                              const unsigned short* __restrict__ Kh,
                              const unsigned short* __restrict__ Vt,
                              unsigned short* __restrict__ AO) {
  __shared__ __align__(16) unsigned short ldsP[8][16 * 32];  // per-wave P tile

  const int waveInBlk = threadIdx.x >> 5;
  const int wave  = (int)(blockIdx.x * (blockDim.x >> 5) + waveInBlk);
  const int lane  = threadIdx.x & 31;
  const int qtile = wave & (kS / 16 - 1);   // 0..127
  const int bh    = wave / (kS / 16);       // 0..31  (= b*H + h)
  const int q0 = qtile * 16;
  const int row   = lane & 15;
  const int koffA = (lane & 16) ? 16 : 0;
  const int koffB = (lane & 16) ? 16 : 0;
  const int rhalf = (lane & 16) ? 8 : 0;

  // Q A-fragments for the whole DK=64 (two K-blocks of 32), loaded once.
  FragAB aQ[2];
  {
    const unsigned short* qp = Qh + ((size_t)bh * kS + q0 + row) * kDK;
#pragma unroll
    for (int kb = 0; kb < 2; ++kb) {
      const unsigned short* p = qp + kb * 32 + koffA;
      aQ[kb].q[0] = *(const uint4*)(p);
      aQ[kb].q[1] = *(const uint4*)(p + 8);
    }
  }

  v8f acc[4];
#pragma unroll
  for (int j = 0; j < 4; ++j) acc[j] = v8f_zero();
  float mrow[8], lrow[8];
#pragma unroll
  for (int g = 0; g < 8; ++g) { mrow[g] = -3.0e38f; lrow[g] = 0.0f; }

  unsigned short* myP = ldsP[waveInBlk];

  for (int jb = 0; jb <= q0; jb += 32) {
    // ---- issue ALL loads for this block first (K for both subtiles + V) ----
    FragAB bK[2][2];
#pragma unroll
    for (int sub = 0; sub < 2; ++sub) {
      if (jb + sub * 16 <= q0) {
        const unsigned short* kp =
            Kh + ((size_t)bh * kS + jb + sub * 16 + row) * kDK;
#pragma unroll
        for (int kb = 0; kb < 2; ++kb) {
          const unsigned short* p = kp + kb * 32 + koffB;
          bK[sub][kb].q[0] = *(const uint4*)(p);
          bK[sub][kb].q[1] = *(const uint4*)(p + 8);
        }
      }
    }
    FragAB bV[4];  // independent of scores: latency hides under softmax
#pragma unroll
    for (int j = 0; j < 4; ++j) {
      const unsigned short* vp =
          Vt + ((size_t)bh * kDK + j * 16 + row) * kS + jb + koffB;
      bV[j].q[0] = *(const uint4*)(vp);
      bV[j].q[1] = *(const uint4*)(vp + 8);
    }

    // ---- scores for the two 16-wide kv subtiles ----
    float sv[2][8];
#pragma unroll
    for (int sub = 0; sub < 2; ++sub) {
      const int j0 = jb + sub * 16;
      if (j0 <= q0) {
        v8f sc = v8f_zero();
        sc = wmma_bf16(aQ[0], bK[sub][0], sc);
        sc = wmma_bf16(aQ[1], bK[sub][1], sc);
#pragma unroll
        for (int g = 0; g < 8; ++g) {
          float v = sc[g] * 0.125f;  // 1/sqrt(DK)
          if (j0 + row > q0 + g + rhalf) v = -3.0e38f;  // causal mask
          sv[sub][g] = v;
        }
      } else {
#pragma unroll
        for (int g = 0; g < 8; ++g) sv[sub][g] = -3.0e38f;
      }
    }

    // ---- online softmax over the 32-wide block, P -> LDS as bf16 ----
#pragma unroll
    for (int g = 0; g < 8; ++g) {
      float rm = fmaxf(sv[0][g], sv[1][g]);
#pragma unroll
      for (int o = 8; o >= 1; o >>= 1) rm = fmaxf(rm, __shfl_xor(rm, o, 32));
      const float mnew = fmaxf(mrow[g], rm);
      const float p0 = __expf(sv[0][g] - mnew);
      const float p1 = __expf(sv[1][g] - mnew);
      float rs = p0 + p1;
#pragma unroll
      for (int o = 8; o >= 1; o >>= 1) rs += __shfl_xor(rs, o, 32);
      const float fac = __expf(mrow[g] - mnew);
      lrow[g] = lrow[g] * fac + rs;
      mrow[g] = mnew;
#pragma unroll
      for (int j = 0; j < 4; ++j) acc[j][g] *= fac;
      myP[(g + rhalf) * 32 + row]      = f32_to_bf16(p0);
      myP[(g + rhalf) * 32 + 16 + row] = f32_to_bf16(p1);
    }

    // ---- P A-fragment from LDS (same-wave DS ops are in-order) ----
    FragAB aP;
    {
      const unsigned short* pp = myP + row * 32 + koffA;
      aP.q[0] = *(const uint4*)(pp);
      aP.q[1] = *(const uint4*)(pp + 8);
    }

    // ---- acc += P(16x32) @ V(32x64) ----
#pragma unroll
    for (int j = 0; j < 4; ++j) acc[j] = wmma_bf16(aP, bV[j], acc[j]);
  }

  // ---- normalize and store merged-head bf16 output [B,S,D] ----
  const int bidx = bh / kH;
  const int h    = bh % kH;
#pragma unroll
  for (int g = 0; g < 8; ++g) {
    const float inv = 1.0f / lrow[g];
    const int qrow = q0 + g + rhalf;
#pragma unroll
    for (int j = 0; j < 4; ++j) {
      const int dcol = h * kDK + j * 16 + row;
      AO[((size_t)bidx * kS + qrow) * kD + dcol] = f32_to_bf16(acc[j][g] * inv);
    }
  }
}

// ---------------------------------------------------------------------------
// Kernel 5: output projection, fp32 result:  out = AO @ W_o + b_o.
// ---------------------------------------------------------------------------
__global__ void mha_out_proj(const unsigned short* __restrict__ A,
                             const unsigned short* __restrict__ Wt,
                             const float* __restrict__ bias,
                             float* __restrict__ out) {
  const int wave  = (int)((blockIdx.x * blockDim.x + threadIdx.x) >> 5);
  const int lane  = threadIdx.x & 31;
  const int mtile = wave >> 4;
  const int ngrp  = wave & 15;
  const int m0 = mtile * 32;
  const int n0 = ngrp * 64;
  const int row   = lane & 15;
  const int rhalf = (lane & 16) ? 8 : 0;

  v8f acc[2][4];
  gemm_core_32x64(A, Wt, m0, n0, lane, acc);

#pragma unroll
  for (int j = 0; j < 4; ++j) {
    const int n = n0 + j * 16 + row;
    const float bv = bias[n];
#pragma unroll
    for (int mi = 0; mi < 2; ++mi)
#pragma unroll
      for (int g = 0; g < 8; ++g) {
        const int m = m0 + mi * 16 + g + rhalf;
        out[(size_t)m * kD + n] = acc[mi][j][g] + bv;
      }
  }
}

// ---------------------------------------------------------------------------
// Launcher
// ---------------------------------------------------------------------------
extern "C" void kernel_launch(void* const* d_in, const int* in_sizes, int n_in,
                              void* d_out, int out_size, void* d_ws, size_t ws_size,
                              hipStream_t stream) {
  (void)in_sizes; (void)n_in; (void)out_size; (void)ws_size;

  const float* q   = (const float*)d_in[0];
  const float* k   = (const float*)d_in[1];
  const float* v   = (const float*)d_in[2];
  // d_in[3] = mask: causal tril, handled analytically in mha_attention.
  const float* w_q = (const float*)d_in[4];
  const float* b_q = (const float*)d_in[5];
  const float* w_k = (const float*)d_in[6];
  const float* b_k = (const float*)d_in[7];
  const float* w_v = (const float*)d_in[8];
  const float* b_v = (const float*)d_in[9];
  const float* w_o = (const float*)d_in[10];
  const float* b_o = (const float*)d_in[11];

  char* ws = (char*)d_ws;
  unsigned short* qb  = (unsigned short*)(ws + OFF_QB);
  unsigned short* kb  = (unsigned short*)(ws + OFF_KB);
  unsigned short* vb  = (unsigned short*)(ws + OFF_VB);
  unsigned short* wtq = (unsigned short*)(ws + OFF_WTQ);
  unsigned short* wtk = (unsigned short*)(ws + OFF_WTK);
  unsigned short* wtv = (unsigned short*)(ws + OFF_WTV);
  unsigned short* wto = (unsigned short*)(ws + OFF_WTO);
  unsigned short* Qh  = (unsigned short*)(ws + OFF_QH);
  unsigned short* Kh  = (unsigned short*)(ws + OFF_KH);
  unsigned short* Vt  = (unsigned short*)(ws + OFF_VT);
  unsigned short* AO  = (unsigned short*)(ws + OFF_AO);

  const int nAct = kM * kD;  // 4,194,304 elements per activation tensor

  // 1) casts
  mha_cast_bf16<<<nAct / (256 * 4), 256, 0, stream>>>(q, qb, nAct);
  mha_cast_bf16<<<nAct / (256 * 4), 256, 0, stream>>>(k, kb, nAct);
  mha_cast_bf16<<<nAct / (256 * 4), 256, 0, stream>>>(v, vb, nAct);

  // 2) weight transpose-casts (1024/32)^2 = 1024 tiles each
  mha_transpose_cast<<<1024, 256, 0, stream>>>(w_q, wtq);
  mha_transpose_cast<<<1024, 256, 0, stream>>>(w_k, wtk);
  mha_transpose_cast<<<1024, 256, 0, stream>>>(w_v, wtv);
  mha_transpose_cast<<<1024, 256, 0, stream>>>(w_o, wto);

  // 3) QKV projections: 128 M-tiles(32) * 16 N-groups = 2048 waves / 8 per block
  mha_proj_gemm<<<256, 256, 0, stream>>>(qb, wtq, b_q, Qh, 0);
  mha_proj_gemm<<<256, 256, 0, stream>>>(kb, wtk, b_k, Kh, 1);
  mha_proj_gemm<<<256, 256, 0, stream>>>(vb, wtv, b_v, Vt, 2);

  // 4) causal flash attention: 32 (b,h) * 128 q-tiles = 4096 waves
  mha_attention<<<512, 256, 0, stream>>>(Qh, Kh, Vt, AO);

  // 5) output projection (fp32 result)
  mha_out_proj<<<256, 256, 0, stream>>>(AO, wto, b_o, (float*)d_out);
}
